// NeighborhoodAttention2D_21388937134551
// MI455X (gfx1250) — matrix operand order repaired and checked by reading, not compiled, not run on hardware
//
#include <hip/hip_runtime.h>
#include <hip/hip_bf16.h>

typedef __attribute__((ext_vector_type(16))) __bf16   v16bf;
typedef __attribute__((ext_vector_type(16))) _Float16 v16h;
typedef __attribute__((ext_vector_type(8)))  float    v8f;

// async-copy pointer types (match clang's builtin prototype: int4 with AS quals)
typedef int v4i_t __attribute__((__vector_size__(16)));
typedef __attribute__((address_space(1))) v4i_t* as1_v4i_p;   // global
typedef __attribute__((address_space(3))) v4i_t* as3_v4i_p;   // LDS

#define DIMC 128
#define NHD  4
#define HDIM 32
#define KSZ  7
#define PADW 3
#define BIMG 8
#define HIMG 56
#define WIMG 56
#define NPIX (BIMG * HIMG * WIMG)          // 25088
#define SCALEQ 0.17677669529663687f        // 32^-0.5

// ---------------------------------------------------------------------------
// Kernel 0: f32 -> bf16 weight conversion (one-time pre-pass into workspace)
// ---------------------------------------------------------------------------
__global__ __launch_bounds__(256)
void cvt_kernel(const float* __restrict__ src, __bf16* __restrict__ dst, int n)
{
    int i = blockIdx.x * 256 + threadIdx.x;
    if (i < n) dst[i] = (__bf16)src[i];
}

// ---------------------------------------------------------------------------
// Kernel 1: qkv = x @ W_qkv^T + b_qkv (q pre-scaled), stored bf16.
// grid = NPIX/64, 256 threads (8 waves). Block: 64 rows x 384 cols.
// Wave w owns col-tiles {3w,3w+1,3w+2} x all 4 row-tiles; B-fragment loaded
// once per (ct,kb) and reused across 4 row-tiles.
// ---------------------------------------------------------------------------
__global__ __launch_bounds__(256)
void qkv_kernel(const float* __restrict__ x, const __bf16* __restrict__ Wb,
                const float* __restrict__ bqkv, __bf16* __restrict__ qkv)
{
    __shared__ __bf16 xs[64 * 136];   // padded stride
    const int blk = blockIdx.x;
    const int tid = threadIdx.x;

    for (int i = tid; i < 64 * 128; i += 256) {
        int r = i >> 7, c = i & 127;
        xs[r * 136 + c] = (__bf16)x[(blk * 64 + r) * 128 + c];
    }
    __syncthreads();

    const int w  = tid >> 5, lane = tid & 31;
    const int hf = lane >> 4, lm = lane & 15;

    v8f acc[4][3] = {};
    for (int kb = 0; kb < 4; ++kb) {
        v16bf a[4];
        #pragma unroll
        for (int rt = 0; rt < 4; ++rt) {
            const __bf16* ap = &xs[(rt * 16 + lm) * 136 + kb * 32 + hf * 8];
            #pragma unroll
            for (int j = 0; j < 8; ++j) { a[rt][j] = ap[j]; a[rt][8 + j] = ap[16 + j]; }
        }
        #pragma unroll
        for (int t = 0; t < 3; ++t) {
            const __bf16* bp = &Wb[((w * 3 + t) * 16 + lm) * 128 + kb * 32 + hf * 8];
            __builtin_prefetch(bp + 32, 0, 0);
            v16bf bfr;
            #pragma unroll
            for (int j = 0; j < 16; ++j) bfr[j] = bp[j < 8 ? j : 8 + j];
            #pragma unroll
            for (int rt = 0; rt < 4; ++rt)
                acc[rt][t] = __builtin_amdgcn_wmma_f32_16x16x32_bf16(
                    false, a[rt], false, bfr, (short)0, acc[rt][t], false, false);
        }
    }
    #pragma unroll
    for (int t = 0; t < 3; ++t) {
        int o = (w * 3 + t) * 16 + lm;
        float bias = bqkv[o];
        float sc   = (o < 128) ? SCALEQ : 1.0f;
        #pragma unroll
        for (int rt = 0; rt < 4; ++rt) {
            #pragma unroll
            for (int r = 0; r < 8; ++r) {
                int row = blk * 64 + rt * 16 + hf * 8 + r;
                qkv[row * 384 + o] = (__bf16)((acc[rt][t][r] + bias) * sc);
            }
        }
    }
}

// ---------------------------------------------------------------------------
// Kernel 2: neighborhood attention per (batch, head, 8x8 pixel tile).
// Union window 14x14 indexed u = wy*16+wx -> padded score matrix 64 x 224.
// QK^T: bf16 WMMA; softmax over 49 valid cols; AV: f16 WMMA.
// K-union staged via async global->LDS copies when available.
// ---------------------------------------------------------------------------
__global__ __launch_bounds__(256)
void attn_kernel(const __bf16* __restrict__ qkv, const float* __restrict__ rpb,
                 __bf16* __restrict__ aout)
{
    __shared__ __bf16   ks[224 * 40];    // K union  [u][d], stride 40
    __shared__ _Float16 vsT[32 * 232];   // V^T union [d][u], stride 232
    __shared__ _Float16 sp[64 * 232];    // scores / probs [pixel][u], stride 232

    const int tile = blockIdx.x;                 // 0..48
    const int h    = blockIdx.y;
    const int b    = blockIdx.z;
    const int ty0  = (tile / 7) * 8;
    const int tx0  = (tile % 7) * 8;
    const int tid  = threadIdx.x;

    // ---- stage K union (async global->LDS 16B copies; zero-fill halo) ----
#if defined(__has_builtin) && __has_builtin(__builtin_amdgcn_global_load_async_to_lds_b128)
  #define KS_ASYNC 1
#else
  #define KS_ASYNC 0
#endif
    for (int chunk = tid; chunk < 224 * 4; chunk += 256) {
        int u = chunk >> 2, d0 = (chunk & 3) * 8;
        int wy = u >> 4, wx = u & 15;
        int py = ty0 + wy - PADW, px = tx0 + wx - PADW;
        __bf16* lp = &ks[u * 40 + d0];
        bool valid = (wx < 14) && ((unsigned)py < 56u) && ((unsigned)px < 56u);
        int g = (b * 56 + py) * 56 + px;
        const __bf16* gp = &qkv[g * 384 + 128 + h * 32 + d0];
#if KS_ASYNC
        if (valid) {
            __builtin_amdgcn_global_load_async_to_lds_b128(
                (as1_v4i_p)gp, (as3_v4i_p)lp, 0, 0);
        } else {
            float4 z = {0.f, 0.f, 0.f, 0.f};
            *(float4*)lp = z;
        }
#else
        float4 val = {0.f, 0.f, 0.f, 0.f};
        if (valid) val = *(const float4*)gp;
        *(float4*)lp = val;
#endif
    }
    // ---- stage V^T union (bf16 -> f16, transposed) — overlaps async K ----
    for (int i = tid; i < 224 * 32; i += 256) {
        int u = i >> 5, d = i & 31;
        int wy = u >> 4, wx = u & 15;
        int py = ty0 + wy - PADW, px = tx0 + wx - PADW;
        _Float16 v = (_Float16)0.f;
        if (wx < 14 && (unsigned)py < 56u && (unsigned)px < 56u) {
            int g = (b * 56 + py) * 56 + px;
            v = (_Float16)(float)qkv[g * 384 + 256 + h * 32 + d];
        }
        vsT[d * 232 + u] = v;
    }
#if KS_ASYNC
  #if defined(__has_builtin) && __has_builtin(__builtin_amdgcn_s_wait_asynccnt)
    __builtin_amdgcn_s_wait_asynccnt(0);
  #else
    asm volatile("s_wait_asynccnt 0x0" ::: "memory");
  #endif
#endif
    __syncthreads();

    const int w  = tid >> 5, lane = tid & 31;
    const int hf = lane >> 4, lm = lane & 15;

    // ---- phase A: S = Q * K_union^T  (64 x 224), bf16 WMMA ----
    {
        const int mtile = w & 3;
        int p  = mtile * 16 + lm;                       // pixel 0..63
        int py = ty0 + (p >> 3), px = tx0 + (p & 7);
        int g  = (b * 56 + py) * 56 + px;
        const __bf16* qp = &qkv[g * 384 + h * 32 + hf * 8];
        v16bf a;
        #pragma unroll
        for (int j = 0; j < 8; ++j) { a[j] = qp[j]; a[8 + j] = qp[16 + j]; }
        for (int nt = (w >> 2); nt < 14; nt += 2) {
            const __bf16* kp = &ks[(nt * 16 + lm) * 40 + hf * 8];
            v16bf bfr;
            #pragma unroll
            for (int j = 0; j < 16; ++j) bfr[j] = kp[j < 8 ? j : 8 + j];
            v8f c = {};
            c = __builtin_amdgcn_wmma_f32_16x16x32_bf16(
                false, a, false, bfr, (short)0, c, false, false);
            #pragma unroll
            for (int r = 0; r < 8; ++r)
                sp[(mtile * 16 + hf * 8 + r) * 232 + nt * 16 + lm] = (_Float16)c[r];
        }
    }
    __syncthreads();

    // ---- phase B: per-pixel softmax over the 49 neighbors (+rpb) ----
    if (tid < 64) {
        int ty = tid >> 3, tx = tid & 7;
        _Float16* row = &sp[tid * 232];
        const float* rp = &rpb[h * 49];
        float mx = -1e30f;
        for (int dy = 0; dy < 7; ++dy)
            for (int dx = 0; dx < 7; ++dx) {
                int u = ((ty + dy) << 4) + tx + dx;
                float s = (float)row[u] + rp[dy * 7 + dx];
                mx = fmaxf(mx, s);
            }
        float sum = 0.f;
        for (int u = 0; u < 224; ++u) {
            int dy = (u >> 4) - ty, dx = (u & 15) - tx;
            float val = 0.f;
            if ((unsigned)dy < 7u && (unsigned)dx < 7u) {
                val = __expf((float)row[u] + rp[dy * 7 + dx] - mx);
                sum += val;
            }
            row[u] = (_Float16)val;   // zero-filled sparse prob row
        }
        float inv = 1.f / sum;
        for (int u = 0; u < 224; ++u)
            row[u] = (_Float16)((float)row[u] * inv);
    }
    __syncthreads();

    // ---- phase C: out = P (64x224, f16) * V_union (224x32, f16) ----
    {
        const int mtile = w & 3;
        const int dt    = w >> 2;     // 0..1 (16-col d-tile)
        v8f c = {};
        for (int kb = 0; kb < 7; ++kb) {
            v16h a;
            const _Float16* ap = &sp[(mtile * 16 + lm) * 232 + kb * 32 + hf * 8];
            #pragma unroll
            for (int j = 0; j < 8; ++j) { a[j] = ap[j]; a[8 + j] = ap[16 + j]; }
            v16h bh;
            const _Float16* vp = &vsT[(dt * 16 + lm) * 232 + kb * 32 + hf * 8];
            #pragma unroll
            for (int j = 0; j < 16; ++j) bh[j] = vp[j < 8 ? j : 8 + j];
            c = __builtin_amdgcn_wmma_f32_16x16x32_f16(
                false, a, false, bh, (short)0, c, false, false);
        }
        #pragma unroll
        for (int r = 0; r < 8; ++r) {
            int p  = mtile * 16 + hf * 8 + r;
            int py = ty0 + (p >> 3), px = tx0 + (p & 7);
            int g  = (b * 56 + py) * 56 + px;
            aout[g * 128 + h * 32 + dt * 16 + lm] = (__bf16)c[r];
        }
    }
}

// ---------------------------------------------------------------------------
// Kernel 3: out = attn_out @ W_proj^T + b_proj (f32 output).
// grid = NPIX/64, 8 waves; wave w owns col-tile w x 4 row-tiles.
// ---------------------------------------------------------------------------
__global__ __launch_bounds__(256)
void proj_kernel(const __bf16* __restrict__ ain, const __bf16* __restrict__ Wb,
                 const float* __restrict__ bp, float* __restrict__ out)
{
    __shared__ __bf16 xs[64 * 136];
    const int blk = blockIdx.x;
    const int tid = threadIdx.x;
    for (int i = tid; i < 64 * 128; i += 256) {
        int r = i >> 7, c = i & 127;
        xs[r * 136 + c] = ain[(blk * 64 + r) * 128 + c];
    }
    __syncthreads();

    const int w  = tid >> 5, lane = tid & 31;
    const int hf = lane >> 4, lm = lane & 15;

    v8f acc[4] = {};
    for (int kb = 0; kb < 4; ++kb) {
        const __bf16* bp2 = &Wb[(w * 16 + lm) * 128 + kb * 32 + hf * 8];
        v16bf bfr;
        #pragma unroll
        for (int j = 0; j < 16; ++j) bfr[j] = bp2[j < 8 ? j : 8 + j];
        #pragma unroll
        for (int rt = 0; rt < 4; ++rt) {
            v16bf a;
            const __bf16* ap = &xs[(rt * 16 + lm) * 136 + kb * 32 + hf * 8];
            #pragma unroll
            for (int j = 0; j < 8; ++j) { a[j] = ap[j]; a[8 + j] = ap[16 + j]; }
            acc[rt] = __builtin_amdgcn_wmma_f32_16x16x32_bf16(
                false, a, false, bfr, (short)0, acc[rt], false, false);
        }
    }
    int o = w * 16 + lm;
    float bias = bp[o];
    #pragma unroll
    for (int rt = 0; rt < 4; ++rt) {
        #pragma unroll
        for (int r = 0; r < 8; ++r)
            out[(blk * 64 + rt * 16 + hf * 8 + r) * 128 + o] = acc[rt][r] + bias;
    }
}

// ---------------------------------------------------------------------------
extern "C" void kernel_launch(void* const* d_in, const int* in_sizes, int n_in,
                              void* d_out, int out_size, void* d_ws, size_t ws_size,
                              hipStream_t stream)
{
    const float* x     = (const float*)d_in[0];
    const float* Wqkv  = (const float*)d_in[1];
    const float* bqkv  = (const float*)d_in[2];
    const float* rpb   = (const float*)d_in[3];
    const float* Wproj = (const float*)d_in[4];
    const float* bproj = (const float*)d_in[5];

    char* ws = (char*)d_ws;
    __bf16* qkv   = (__bf16*)ws;                      // NPIX*384 bf16
    __bf16* aout  = (__bf16*)(ws + (size_t)NPIX * 384 * 2);
    __bf16* wqkvb = (__bf16*)(ws + (size_t)NPIX * 384 * 2 + (size_t)NPIX * 128 * 2);
    __bf16* wprjb = wqkvb + 384 * 128;

    cvt_kernel<<<(384 * 128 + 255) / 256, 256, 0, stream>>>(Wqkv, wqkvb, 384 * 128);
    cvt_kernel<<<(128 * 128 + 255) / 256, 256, 0, stream>>>(Wproj, wprjb, 128 * 128);
    qkv_kernel<<<NPIX / 64, 256, 0, stream>>>(x, wqkvb, bqkv, qkv);
    attn_kernel<<<dim3(49, NHD, BIMG), 256, 0, stream>>>(qkv, rpb, aout);
    proj_kernel<<<NPIX / 64, 256, 0, stream>>>(aout, wprjb, bproj, (float*)d_out);
}
